// PPGN_7842610283371
// MI455X (gfx1250) — matrix operand order repaired and checked by reading, not compiled, not run on hardware
//
#include <hip/hip_runtime.h>
#include <math.h>

// ---------------------------------------------------------------------------
// PPGN forward for MI455X (gfx1250, wave32, WMMA).
// All matmuls run through v_wmma_f32_16x16x32_bf16 (bf16 in, f32 accum).
// Activations stored bf16 (B, 416, 64, 64) with channels padded 400->416.
// GEMM: per-wave 16x32 output tile, X staged via coalesced b128 loads into a
// per-wave LDS slab (same-wave DS ordering -> no barriers needed).
// ---------------------------------------------------------------------------

#define DEVINL __device__ __forceinline__

typedef __attribute__((ext_vector_type(16))) __bf16 v16bf;
typedef __attribute__((ext_vector_type(8)))  float  v8f;

static constexpr int BN   = 8;     // batches
static constexpr int NN   = 4096;  // 64*64 spatial
static constexpr int HC   = 400;   // hidden channels
static constexpr int KC   = 416;   // padded channel stride (mult of 32)
static constexpr int KC0  = 32;    // padded input channels (CIN=22 -> 32)

DEVINL __bf16 us2bf(unsigned short u){ union { unsigned short s; __bf16 b; } c; c.s = u; return c.b; }
DEVINL unsigned short f2bf(float f){
  union { float f; unsigned int u; } c; c.f = f;
  unsigned int r = c.u + 0x7FFFu + ((c.u >> 16) & 1u);
  return (unsigned short)(r >> 16);
}
DEVINL float bf2f(unsigned short u){ union { unsigned int u; float f; } c; c.u = ((unsigned int)u) << 16; return c.f; }

// ---------------------------------------------------------------------------
// Fragment-gather accumulate (used by the per-channel 64x64 matmuls):
// acc += A(16xKp) * B(Kpx16), A rows contiguous in k, B at xcol[k*ldx].
// Layouts per CDNA5 ISA 7.12.2.
// ---------------------------------------------------------------------------
DEVINL void wmma_k_accum(const unsigned short* __restrict__ wrow,
                         const unsigned short* __restrict__ xcol,
                         int Kp, int ldx, int lane, v8f& acc)
{
  const int khA = ((lane >> 4) & 1) * 8;
  const int khB = ((lane >> 4) & 1) * 16;
  for (int k0 = 0; k0 < Kp; k0 += 32) {
    v16bf a, bm;
#pragma unroll
    for (int v = 0; v < 8; ++v) {
      int ka = k0 + ((v & 4) << 2) + khA + ((v & 3) << 1);
      unsigned int pr = *(const unsigned int*)(wrow + ka);
      a[2 * v]     = us2bf((unsigned short)(pr & 0xFFFFu));
      a[2 * v + 1] = us2bf((unsigned short)(pr >> 16));
      int kb = k0 + ((v & 4) << 2) + khA + ((v & 3) << 1);
      bm[2 * v]     = us2bf(xcol[(size_t)kb * ldx]);
      bm[2 * v + 1] = us2bf(xcol[(size_t)(kb + 1) * ldx]);
    }
    acc = __builtin_amdgcn_wmma_f32_16x16x32_bf16(false, a, false, bm,
                                                  (short)0, acc, false, false);
  }
}

// ---------------------------------------------------------------------------
// GEMM inner loop over one (W, X) source with LDS-staged X tiles.
// Wave computes a 16x32 output tile: A frag shared across two WMMAs.
//  - xt: per-wave LDS slab, 32(K) x 32(N) bf16 (2 KB)
// Same-wave DS ops are processed in order -> store-then-load needs no barrier.
// ---------------------------------------------------------------------------
DEVINL void gemm_tiles(const unsigned short* __restrict__ W, int Kp,
                       const unsigned short* __restrict__ X, int KCx,
                       int b, int p0, int to, int lane,
                       unsigned short* __restrict__ xt,
                       v8f& acc0, v8f& acc1)
{
  const int khA = ((lane >> 4) & 1) * 8;
  const int khB = ((lane >> 4) & 1) * 16;
  const int n   = lane & 15;
  const unsigned short* wrow  = W + (size_t)(to * 16 + n) * Kp;
  const unsigned short* xbase = X + (size_t)b * KCx * NN + p0;

  for (int k0 = 0; k0 < Kp; k0 += 32) {
    // --- stage X chunk: 32 rows (k) x 32 cols, lane -> row k0+lane ---
    const unsigned short* xrow = xbase + (size_t)(k0 + lane) * NN;
    uint4 d0 = ((const uint4*)xrow)[0];   // cols  0..7
    uint4 d1 = ((const uint4*)xrow)[1];   // cols  8..15
    uint4 d2 = ((const uint4*)xrow)[2];   // cols 16..23
    uint4 d3 = ((const uint4*)xrow)[3];   // cols 24..31
    uint4* dst = (uint4*)(xt + lane * 32);
    dst[0] = d0; dst[1] = d1; dst[2] = d2; dst[3] = d3;

    // --- A fragment from weight row (contiguous k pairs) ---
    v16bf a;
#pragma unroll
    for (int v = 0; v < 8; ++v) {
      int ka = k0 + ((v & 4) << 2) + khA + ((v & 3) << 1);
      unsigned int pr = *(const unsigned int*)(wrow + ka);
      a[2 * v]     = us2bf((unsigned short)(pr & 0xFFFFu));
      a[2 * v + 1] = us2bf((unsigned short)(pr >> 16));
    }
    // --- two B fragments from LDS tile (row-major 32x32) ---
    v16bf b0, b1;
#pragma unroll
    for (int v = 0; v < 8; ++v) {
      int kb = ((v & 4) << 2) + khA + ((v & 3) << 1);
      b0[2 * v]     = us2bf(xt[kb * 32 + n]);
      b0[2 * v + 1] = us2bf(xt[(kb + 1) * 32 + n]);
      b1[2 * v]     = us2bf(xt[kb * 32 + 16 + n]);
      b1[2 * v + 1] = us2bf(xt[(kb + 1) * 32 + 16 + n]);
    }
    acc0 = __builtin_amdgcn_wmma_f32_16x16x32_bf16(false, a, false, b0,
                                                   (short)0, acc0, false, false);
    acc1 = __builtin_amdgcn_wmma_f32_16x16x32_bf16(false, a, false, b1,
                                                   (short)0, acc1, false, false);
  }
}

// ---------------------------------------------------------------------------
// Generic 1x1-conv GEMM: Out[b,o,p] = act( sum_k W1[o,k] X1[b,k,p]
//                                        + sum_k W2[o,k] X2[b,k,p] + bias[o] )
// One wave per 16x32 tile. Columns = B*4096 (32-col tiles never cross batch).
// ---------------------------------------------------------------------------
__global__ void wmma_gemm_kernel(const unsigned short* __restrict__ W1, int Kp1,
                                 const unsigned short* __restrict__ X1, int KC1,
                                 const unsigned short* __restrict__ W2, int Kp2,
                                 const unsigned short* __restrict__ X2, int KC2,
                                 const float* __restrict__ bias,
                                 unsigned short* __restrict__ Out, int OC,
                                 int nTO, int doRelu)
{
  __shared__ __align__(16) unsigned short xs[8 * 1024];  // 2 KB per wave
  int w    = threadIdx.x >> 5;
  int lane = threadIdx.x & 31;
  int wid  = blockIdx.x * 8 + w;
  int to = wid % nTO;          // output-channel tile
  int tq = wid / nTO;          // 32-wide column tile (0..1023)
  int b  = tq >> 7;            // 128 column tiles per batch
  int p0 = (tq & 127) << 5;
  int n  = lane & 15;
  int half8 = ((lane >> 4) & 1) * 8;
  unsigned short* xt = xs + w * 1024;

  v8f acc0 = {0.f, 0.f, 0.f, 0.f, 0.f, 0.f, 0.f, 0.f};
  v8f acc1 = {0.f, 0.f, 0.f, 0.f, 0.f, 0.f, 0.f, 0.f};
  gemm_tiles(W1, Kp1, X1, KC1, b, p0, to, lane, xt, acc0, acc1);
  if (W2) gemm_tiles(W2, Kp2, X2, KC2, b, p0, to, lane, xt, acc0, acc1);

#pragma unroll
  for (int v = 0; v < 8; ++v) {
    int m = to * 16 + half8 + v;
    float r0 = acc0[v] + bias[m];
    float r1 = acc1[v] + bias[m];
    if (doRelu) { r0 = fmaxf(r0, 0.f); r1 = fmaxf(r1, 0.f); }
    size_t o = ((size_t)(b * OC + m)) * NN + p0 + n;
    Out[o]      = f2bf(r0);
    Out[o + 16] = f2bf(r1);
  }
}

// ---------------------------------------------------------------------------
// Per-channel 64x64x64 matmul: mult[b,c] = m1[b,c] @ m2[b,c]. 16 tiles per
// (b,c); K=64 -> 2 WMMAs per tile.
// ---------------------------------------------------------------------------
__global__ void mult_kernel(const unsigned short* __restrict__ M1,
                            const unsigned short* __restrict__ M2,
                            unsigned short* __restrict__ Out)
{
  int wid  = blockIdx.x * (blockDim.x >> 5) + (threadIdx.x >> 5);
  int lane = threadIdx.x & 31;
  int bc = wid >> 4;                     // 0..3199
  if (bc >= BN * HC) return;
  int t = wid & 15, tm = t >> 2, tn = t & 3;
  int b = bc / HC, c = bc % HC;
  size_t base = (size_t)(b * KC + c) * NN;
  int n = lane & 15;
  int half8 = ((lane >> 4) & 1) * 8;

  v8f acc = {0.f, 0.f, 0.f, 0.f, 0.f, 0.f, 0.f, 0.f};
  const unsigned short* arow = M1 + base + (size_t)(tm * 16 + n) * 64;
  const unsigned short* xcol = M2 + base + tn * 16 + n;
  wmma_k_accum(arow, xcol, 64, 64, lane, acc);
#pragma unroll
  for (int v = 0; v < 8; ++v)
    Out[base + (size_t)(tm * 16 + half8 + v) * 64 + tn * 16 + n] = f2bf(acc[v]);
}

// ---------------------------------------------------------------------------
// Build z0 (B, 32, 64, 64) bf16: edge scatter (via per-(b,m,n) edge scan ->
// deterministic, no atomics), pairwise distance, diag-embedded node features,
// zero pad channels 22..31.
// ---------------------------------------------------------------------------
__global__ void build_z0_kernel(const float* __restrict__ x,
                                const float* __restrict__ pos,
                                const float* __restrict__ edge_attr,
                                const int*   __restrict__ eidx,  // (2, 4096)
                                unsigned short* __restrict__ z0)
{
  int gid = blockIdx.x * blockDim.x + threadIdx.x;
  if (gid >= BN * NN) return;
  int b = gid >> 12;
  int r = gid & 4095;
  int m = r >> 6;
  int n = r & 63;

  // z[b,c,m,n] = dense_edge[b, n, m, c]: edges with src_local==n, dst_local==m
  float e0 = 0.f, e1 = 0.f, e2 = 0.f, e3 = 0.f, e4 = 0.f;
  const int* srcv = eidx;
  const int* dstv = eidx + 4096;
  int ebase = b * 512;
  for (int t = 0; t < 512; ++t) {
    int e = ebase + t;
    int s = srcv[e] - b * 64;
    int d = dstv[e] - b * 64;
    if (s == n && d == m) {
      const float* ea = edge_attr + (size_t)e * 5;
      e0 += ea[0]; e1 += ea[1]; e2 += ea[2]; e3 += ea[3]; e4 += 1.0f;
    }
  }
  const float* pn = pos + (size_t)(b * 64 + n) * 3;
  const float* pm = pos + (size_t)(b * 64 + m) * 3;
  float dx = pn[0] - pm[0], dy = pn[1] - pm[1], dz = pn[2] - pm[2];
  float d2 = dx * dx + dy * dy + dz * dz;
  float dist = (d2 > 0.f) ? sqrtf(d2) : 0.f;

  size_t base = (size_t)b * KC0 * NN + r;
  z0[base + 0 * NN] = f2bf(e0);
  z0[base + 1 * NN] = f2bf(e1);
  z0[base + 2 * NN] = f2bf(e2);
  z0[base + 3 * NN] = f2bf(e3);
  z0[base + 4 * NN] = f2bf(e4);
  z0[base + 5 * NN] = f2bf(dist);
#pragma unroll
  for (int c = 0; c < 16; ++c) {
    float v = (m == n) ? x[(size_t)(b * 64 + n) * 16 + c] : 0.f;
    z0[base + (size_t)(6 + c) * NN] = f2bf(v);
  }
  for (int c = 22; c < 32; ++c) z0[base + (size_t)c * NN] = 0;
}

// Convert f32 weight slice (O x ncols, row stride srcLd, col offset col0)
// to bf16 (O x Kpad), zero-padding k >= ncols.
__global__ void cvt_w_kernel(const float* __restrict__ src, int srcLd, int col0,
                             int ncols, unsigned short* __restrict__ dst,
                             int Kpad, int O)
{
  int i = blockIdx.x * blockDim.x + threadIdx.x;
  if (i >= O * Kpad) return;
  int o = i / Kpad, k = i % Kpad;
  float v = (k < ncols) ? src[(size_t)o * srcLd + col0 + k] : 0.f;
  dst[i] = f2bf(v);
}

// Zero pad rows [rowsUsed, KCc) of a (BN, KCc, 4096) bf16 activation buffer.
__global__ void zero_pad_kernel(unsigned short* __restrict__ p, int KCc, int rowsUsed)
{
  int per = (KCc - rowsUsed) * NN;
  int total = BN * per;
  for (int i = blockIdx.x * blockDim.x + threadIdx.x; i < total;
       i += gridDim.x * blockDim.x) {
    int b = i / per, r = i % per;
    p[(size_t)(b * KCc + rowsUsed) * NN + r] = 0;
  }
}

// ---------------------------------------------------------------------------
// Pooling stage 1: per (b,c) -> max over diagonal, max over off-diagonal,
// min over all (z is bf16 (BN, KC, 64, 64)).
// ---------------------------------------------------------------------------
__global__ void pool_stage1(const unsigned short* __restrict__ Z,
                            float* __restrict__ maxdiag,
                            float* __restrict__ maxoff,
                            float* __restrict__ minall)
{
  int bc = blockIdx.x;
  int b = bc / HC, c = bc % HC;
  size_t base = (size_t)(b * KC + c) * NN;
  int tid = threadIdx.x;  // 64 threads
  float dg = -3.4e38f, mo = -3.4e38f, mn = 3.4e38f;
  for (int i = tid; i < NN; i += 64) {
    float v = bf2f(Z[base + i]);
    int mm = i >> 6, nn2 = i & 63;
    if (mm == nn2) dg = fmaxf(dg, v); else mo = fmaxf(mo, v);
    mn = fminf(mn, v);
  }
  __shared__ float sd[64], so[64], sn[64];
  sd[tid] = dg; so[tid] = mo; sn[tid] = mn;
  __syncthreads();
  for (int s = 32; s > 0; s >>= 1) {
    if (tid < s) {
      sd[tid] = fmaxf(sd[tid], sd[tid + s]);
      so[tid] = fmaxf(so[tid], so[tid + s]);
      sn[tid] = fminf(sn[tid], sn[tid + s]);
    }
    __syncthreads();
  }
  if (tid == 0) { maxdiag[bc] = sd[0]; maxoff[bc] = so[0]; minall[bc] = sn[0]; }
}

// Stage 2: val = |max(maxdiag) + max(-z)| = |gmax_diag - gmin|;
// feat[b, c] = maxdiag, feat[b, 400+c] = max(maxoff_nondiag, maxdiag - val).
__global__ void pool_stage2(const float* __restrict__ maxdiag,
                            const float* __restrict__ maxoff,
                            const float* __restrict__ minall,
                            float* __restrict__ feat)
{
  __shared__ float sd[256], sn[256];
  __shared__ float sval;
  int tid = threadIdx.x;
  float gd = -3.4e38f, gm = 3.4e38f;
  for (int i = tid; i < BN * HC; i += 256) {
    gd = fmaxf(gd, maxdiag[i]);
    gm = fminf(gm, minall[i]);
  }
  sd[tid] = gd; sn[tid] = gm;
  __syncthreads();
  for (int s = 128; s > 0; s >>= 1) {
    if (tid < s) {
      sd[tid] = fmaxf(sd[tid], sd[tid + s]);
      sn[tid] = fminf(sn[tid], sn[tid + s]);
    }
    __syncthreads();
  }
  if (tid == 0) sval = fabsf(sd[0] - sn[0]);
  __syncthreads();
  float val = sval;
  for (int i = tid; i < BN * HC; i += 256) {
    int b = i / HC, c = i % HC;
    feat[b * 800 + c]       = maxdiag[i];
    feat[b * 800 + 400 + c] = fmaxf(maxoff[i], maxdiag[i] - val);
  }
}

// FC head (f32, single block, LDS-resident): 800 -> 512 -> 256 -> 1 per batch.
__global__ void fc_head_kernel(const float* __restrict__ feat,
                               const float* __restrict__ w1, const float* __restrict__ b1,
                               const float* __restrict__ w2, const float* __restrict__ b2,
                               const float* __restrict__ w3, const float* __restrict__ b3,
                               float* __restrict__ out)
{
  __shared__ float sf[BN * 800];
  __shared__ float h1[BN * 512];
  __shared__ float h2[BN * 256];
  int tid = threadIdx.x;
  for (int i = tid; i < BN * 800; i += 256) sf[i] = feat[i];
  __syncthreads();
  for (int i = tid; i < BN * 512; i += 256) {
    int b = i >> 9, o = i & 511;
    float s = b1[o];
    const float* wr = w1 + (size_t)o * 800;
    const float* fr = sf + b * 800;
    for (int c = 0; c < 800; ++c) s += wr[c] * fr[c];
    h1[i] = fmaxf(s, 0.f);
  }
  __syncthreads();
  for (int i = tid; i < BN * 256; i += 256) {
    int b = i >> 8, o = i & 255;
    float s = b2[o];
    const float* wr = w2 + (size_t)o * 512;
    const float* hr = h1 + b * 512;
    for (int c = 0; c < 512; ++c) s += wr[c] * hr[c];
    h2[i] = fmaxf(s, 0.f);
  }
  __syncthreads();
  if (tid < BN) {
    float s = b3[0];
    const float* hr = h2 + tid * 256;
    for (int c = 0; c < 256; ++c) s += w3[c] * hr[c];
    out[tid] = s;
  }
}

// ---------------------------------------------------------------------------
// Host side
// ---------------------------------------------------------------------------
static void* ws_alloc(char*& cur, size_t bytes)
{
  void* p = (void*)cur;
  cur += (bytes + 255) & ~(size_t)255;
  return p;
}

static void launch_gemm(const unsigned short* W1, int Kp1, const unsigned short* X1, int KC1,
                        const unsigned short* W2, int Kp2, const unsigned short* X2, int KC2,
                        const float* bias, unsigned short* Out, int OC, int relu,
                        hipStream_t s)
{
  const int nTO = HC / 16;                  // 25 output-channel tiles
  const int waves = nTO * (BN * NN / 32);   // 25 * 1024 (16x32 tiles)
  wmma_gemm_kernel<<<waves / 8, 256, 0, s>>>(W1, Kp1, X1, KC1, W2, Kp2, X2, KC2,
                                             bias, Out, OC, nTO, relu);
}

static void launch_cvt(const float* src, int srcLd, int col0, int ncols,
                       unsigned short* dst, int Kpad, hipStream_t s)
{
  int total = HC * Kpad;
  cvt_w_kernel<<<(total + 255) / 256, 256, 0, s>>>(src, srcLd, col0, ncols, dst, Kpad, HC);
}

extern "C" void kernel_launch(void* const* d_in, const int* in_sizes, int n_in,
                              void* d_out, int out_size, void* d_ws, size_t ws_size,
                              hipStream_t stream)
{
  (void)in_sizes; (void)n_in; (void)out_size; (void)ws_size;

  // Input order: x, pos, edge_attr, edge_index, batch, then params pytree
  // (dict keys sorted): fc1(w,b), fc2(w,b), fc3(w,b),
  // rb{1,2,3}: m1(w0,b0,w1,b1), m2(w0,b0,w1,b1), skip(w,b).
  const float* x    = (const float*)d_in[0];
  const float* pos  = (const float*)d_in[1];
  const float* eatt = (const float*)d_in[2];
  const int*   eidx = (const int*)d_in[3];
  const float* fc1w = (const float*)d_in[5];
  const float* fc1b = (const float*)d_in[6];
  const float* fc2w = (const float*)d_in[7];
  const float* fc2b = (const float*)d_in[8];
  const float* fc3w = (const float*)d_in[9];
  const float* fc3b = (const float*)d_in[10];
  const int rbBase[3] = {11, 21, 31};

  char* cur = (char*)d_ws;
  const size_t actEl = (size_t)BN * KC * NN;
  unsigned short* z0 = (unsigned short*)ws_alloc(cur, (size_t)BN * KC0 * NN * 2);
  unsigned short* za = (unsigned short*)ws_alloc(cur, actEl * 2);
  unsigned short* zb = (unsigned short*)ws_alloc(cur, actEl * 2);
  unsigned short* t1 = (unsigned short*)ws_alloc(cur, actEl * 2);
  unsigned short* t2 = (unsigned short*)ws_alloc(cur, actEl * 2);
  unsigned short* t3 = (unsigned short*)ws_alloc(cur, actEl * 2);

  // bf16 weights (K padded to /32). Per block: m1a, m1b, m2a, m2b, sk1, sk2.
  unsigned short* wm1a[3]; unsigned short* wm1b[3];
  unsigned short* wm2a[3]; unsigned short* wm2b[3];
  unsigned short* wsk1[3]; unsigned short* wsk2[3];
  const int KpIn[3] = {KC0, KC, KC};   // first-layer / skip-slice-1 padded K
  for (int i = 0; i < 3; ++i) {
    wm1a[i] = (unsigned short*)ws_alloc(cur, (size_t)HC * KpIn[i] * 2);
    wm1b[i] = (unsigned short*)ws_alloc(cur, (size_t)HC * KC * 2);
    wm2a[i] = (unsigned short*)ws_alloc(cur, (size_t)HC * KpIn[i] * 2);
    wm2b[i] = (unsigned short*)ws_alloc(cur, (size_t)HC * KC * 2);
    wsk1[i] = (unsigned short*)ws_alloc(cur, (size_t)HC * KpIn[i] * 2);
    wsk2[i] = (unsigned short*)ws_alloc(cur, (size_t)HC * KC * 2);
  }
  float* rdiag = (float*)ws_alloc(cur, BN * HC * 4);
  float* roff  = (float*)ws_alloc(cur, BN * HC * 4);
  float* rmin  = (float*)ws_alloc(cur, BN * HC * 4);
  float* feat  = (float*)ws_alloc(cur, BN * 800 * 4);

  // --- weight conversion ---
  for (int i = 0; i < 3; ++i) {
    int base   = rbBase[i];
    int cin    = (i == 0) ? 22 : HC;          // real input channels
    int skLd   = cin + HC;                    // 422 or 800
    launch_cvt((const float*)d_in[base + 0], cin,  0,   cin, wm1a[i], KpIn[i], stream);
    launch_cvt((const float*)d_in[base + 2], HC,   0,   HC,  wm1b[i], KC,      stream);
    launch_cvt((const float*)d_in[base + 4], cin,  0,   cin, wm2a[i], KpIn[i], stream);
    launch_cvt((const float*)d_in[base + 6], HC,   0,   HC,  wm2b[i], KC,      stream);
    launch_cvt((const float*)d_in[base + 8], skLd, 0,   cin, wsk1[i], KpIn[i], stream);
    launch_cvt((const float*)d_in[base + 8], skLd, cin, HC,  wsk2[i], KC,      stream);
  }

  // --- zero channel-pad rows of activation buffers (400..415) ---
  zero_pad_kernel<<<2048, 256, 0, stream>>>(za, KC, HC);
  zero_pad_kernel<<<2048, 256, 0, stream>>>(zb, KC, HC);
  zero_pad_kernel<<<2048, 256, 0, stream>>>(t1, KC, HC);
  zero_pad_kernel<<<2048, 256, 0, stream>>>(t2, KC, HC);
  zero_pad_kernel<<<2048, 256, 0, stream>>>(t3, KC, HC);

  // --- build z0 (B, 32, 64, 64) ---
  build_z0_kernel<<<(BN * NN) / 256, 256, 0, stream>>>(x, pos, eatt, eidx, z0);

  // --- three regular blocks ---
  const unsigned short* zin = z0;
  int kcin = KC0;
  unsigned short* zouts[3] = {za, zb, za};
  for (int i = 0; i < 3; ++i) {
    const float* bm1a = (const float*)d_in[rbBase[i] + 1];
    const float* bm1b = (const float*)d_in[rbBase[i] + 3];
    const float* bm2a = (const float*)d_in[rbBase[i] + 5];
    const float* bm2b = (const float*)d_in[rbBase[i] + 7];
    const float* bsk  = (const float*)d_in[rbBase[i] + 9];
    // m1 = relu(conv2(relu(conv1(z))))
    launch_gemm(wm1a[i], KpIn[i], zin, kcin, nullptr, 0, nullptr, 0, bm1a, t3, KC, 1, stream);
    launch_gemm(wm1b[i], KC,      t3,  KC,   nullptr, 0, nullptr, 0, bm1b, t1, KC, 1, stream);
    // m2
    launch_gemm(wm2a[i], KpIn[i], zin, kcin, nullptr, 0, nullptr, 0, bm2a, t3, KC, 1, stream);
    launch_gemm(wm2b[i], KC,      t3,  KC,   nullptr, 0, nullptr, 0, bm2b, t2, KC, 1, stream);
    // per-channel 64x64 matmul: t3 = t1 @ t2
    mult_kernel<<<(BN * HC * 16) / 8, 256, 0, stream>>>(t1, t2, t3);
    // skip conv over concat([z, mult]) = Wsk1*z + Wsk2*mult + b  (no relu)
    launch_gemm(wsk1[i], KpIn[i], zin, kcin, wsk2[i], KC, t3, KC, bsk, zouts[i], KC, 0, stream);
    zin = zouts[i];
    kcin = KC;
  }
  unsigned short* zfin = zouts[2];  // za

  // --- diag/offdiag maxpool + MLP head ---
  pool_stage1<<<BN * HC, 64, 0, stream>>>(zfin, rdiag, roff, rmin);
  pool_stage2<<<1, 256, 0, stream>>>(rdiag, roff, rmin, feat);
  fc_head_kernel<<<1, 256, 0, stream>>>(feat, fc1w, fc1b, fc2w, fc2b, fc3w, fc3b,
                                        (float*)d_out);
}
